// MiniTransformer_48498770707316
// MI455X (gfx1250) — matrix operand order repaired and checked by reading, not compiled, not run on hardware
//
#include <hip/hip_runtime.h>
#include <hip/hip_bf16.h>
#include <math.h>

#define D_MODEL 512
#define D_FF    2048
#define SEQ     2048
#define BATCH   8
#define MTOT    (BATCH * SEQ)   // 16384 rows total

typedef __attribute__((ext_vector_type(16))) __bf16       v16bf;
typedef __attribute__((ext_vector_type(8)))  float        v8f;
typedef __attribute__((ext_vector_type(4)))  unsigned int u32x4;
typedef __hip_bfloat16 bf16_t;

// One WMMA 16-bit fragment: 16 bf16 values per lane = 2 x 16-byte loads
union Frag16 { v16bf v; u32x4 q[2]; };

__device__ __forceinline__ v8f wmma_bf16(v16bf a, v16bf b, v8f c) {
  // v_wmma_f32_16x16x32_bf16 : D = A(16x32) * B(32x16) + C(16x16 f32)
  return __builtin_amdgcn_wmma_f32_16x16x32_bf16(
      false, a, false, b, (short)0, c, false, false);
}

// A fragment (16x32, row-major source, this lane = row m0+(lane&15)):
// elements 0..7  <- K = g*8 + 0..7 ;  elements 8..15 <- K = 16 + g*8 + 0..7
__device__ __forceinline__ void load_afrag(Frag16& f, const bf16_t* aRow, int k, int g) {
  f.q[0] = *reinterpret_cast<const u32x4*>(aRow + k + g * 8);
  f.q[1] = *reinterpret_cast<const u32x4*>(aRow + k + 16 + g * 8);
}
// B fragment from Bt (row-major [N,K]; this lane = col n0+(lane&15)):
// elements 0..7  <- K = g*8 + 0..7 ;  elements 8..15 <- K = 16 + g*8 + 0..7
__device__ __forceinline__ void load_bfrag(Frag16& f, const bf16_t* bCol, int k, int g) {
  f.q[0] = *reinterpret_cast<const u32x4*>(bCol + k + g * 8);
  f.q[1] = *reinterpret_cast<const u32x4*>(bCol + k + 16 + g * 8);
}

// ---------------------------------------------------------------------------
// Generic WMMA GEMM:  C[M,N] = act(A[M,K] @ Bt[N,K]^T * scale + bias) + res
// Block tile 64(M) x 256(N): 8 waves (2 in M x 4 in N); each wave 32x64
// (2 A-frags x 4 B-frags -> 8 WMMAs per k-step, 12 b128 loads: ~21 FLOP/B).
// OUTMODE: 0 = fp32 store, 1 = bf16 store, 2 = bf16 store V-transposed
// ---------------------------------------------------------------------------
template <bool BIAS, bool RES, bool GELU_ACT, int OUTMODE>
__global__ __launch_bounds__(256) void gemm_wmma_kernel(
    const bf16_t* __restrict__ A, long long strideA, int lda,
    const bf16_t* __restrict__ Bt, long long strideB, int ldb,
    const float* __restrict__ bias, const float* __restrict__ res,
    void* __restrict__ out, long long strideOut,
    int K, int N, float scaleOut) {
  const int bz = blockIdx.z;
  A  += (long long)bz * strideA;
  Bt += (long long)bz * strideB;

  const int w    = threadIdx.x >> 5;
  const int lane = threadIdx.x & 31;
  const int g    = lane >> 4;
  const int c    = lane & 15;
  const int m0   = blockIdx.y * 64 + (w & 1) * 32;
  const int n0   = blockIdx.x * 256 + (w >> 1) * 64;

  const bf16_t* aRow0 = A + (long long)(m0 + c) * lda;
  const bf16_t* aRow1 = aRow0 + (long long)16 * lda;
  const bf16_t* bCol0 = Bt + (long long)(n0 + c) * ldb;
  const bf16_t* bCol1 = bCol0 + (long long)16 * ldb;
  const bf16_t* bCol2 = bCol0 + (long long)32 * ldb;
  const bf16_t* bCol3 = bCol0 + (long long)48 * ldb;

  v8f acc[2][4] = {};
  for (int k = 0; k < K; k += 32) {
    if (k + 128 < K) {  // global_prefetch_b8, 256B lookahead on streamed rows
      __builtin_prefetch(aRow0 + k + 128, 0, 1);
      __builtin_prefetch(aRow1 + k + 128, 0, 1);
      __builtin_prefetch(bCol0 + k + 128, 0, 1);
      __builtin_prefetch(bCol1 + k + 128, 0, 1);
      __builtin_prefetch(bCol2 + k + 128, 0, 1);
      __builtin_prefetch(bCol3 + k + 128, 0, 1);
    }
    Frag16 a0, a1;
    load_afrag(a0, aRow0, k, g);
    load_afrag(a1, aRow1, k, g);
    Frag16 bf;
    load_bfrag(bf, bCol0, k, g);
    acc[0][0] = wmma_bf16(a0.v, bf.v, acc[0][0]);
    acc[1][0] = wmma_bf16(a1.v, bf.v, acc[1][0]);
    load_bfrag(bf, bCol1, k, g);
    acc[0][1] = wmma_bf16(a0.v, bf.v, acc[0][1]);
    acc[1][1] = wmma_bf16(a1.v, bf.v, acc[1][1]);
    load_bfrag(bf, bCol2, k, g);
    acc[0][2] = wmma_bf16(a0.v, bf.v, acc[0][2]);
    acc[1][2] = wmma_bf16(a1.v, bf.v, acc[1][2]);
    load_bfrag(bf, bCol3, k, g);
    acc[0][3] = wmma_bf16(a0.v, bf.v, acc[0][3]);
    acc[1][3] = wmma_bf16(a1.v, bf.v, acc[1][3]);
  }

#pragma unroll
  for (int mi = 0; mi < 2; ++mi) {
#pragma unroll
    for (int t = 0; t < 4; ++t) {
      const int nn = n0 + t * 16 + c;
#pragma unroll
      for (int j = 0; j < 8; ++j) {
        const int mm = m0 + mi * 16 + g * 8 + j;  // D layout: vgpr j -> row g*8+j
        float val = acc[mi][t][j] * scaleOut;
        if (BIAS) val += bias[nn];
        if (GELU_ACT) val = 0.5f * val * (1.0f + erff(val * 0.70710678118654752f));
        if (RES) val += res[(long long)mm * N + nn];
        if (OUTMODE == 0) {
          ((float*)out)[bz * strideOut + (long long)mm * N + nn] = val;
        } else if (OUTMODE == 1) {
          ((bf16_t*)out)[bz * strideOut + (long long)mm * N + nn] = __float2bfloat16(val);
        } else {  // V transposed: out[batch][n][seq]
          const int bb = mm >> 11;        // global row / SEQ
          const int sr = mm & (SEQ - 1);  // row within batch
          ((bf16_t*)out)[((long long)bb * N + nn) * (long long)SEQ + sr] =
              __float2bfloat16(val);
        }
      }
    }
  }
}

// ---------------------------------------------------------------------------
// Fused attention scores + exact softmax -> P (bf16).
// One block = 16 query rows x full 2048 keys. Scores staged in 128KB LDS.
// Q is pre-scaled by 1/sqrt(D). 8 waves; wave w owns key range [w*256, w*256+256).
// Q tile is register-resident (16 A-fragments): 2 loads per WMMA.
// ---------------------------------------------------------------------------
__global__ __launch_bounds__(256) void attn_scores_softmax_kernel(
    const bf16_t* __restrict__ Q, const bf16_t* __restrict__ Kmat,
    bf16_t* __restrict__ P) {
  __shared__ float sS[16 * SEQ];  // 128 KB

  const int blk    = blockIdx.x;             // 16 rows per block
  const int batch  = (blk * 16) >> 11;
  const int rowInB = (blk * 16) & (SEQ - 1);
  const bf16_t* Qb = Q    + (long long)batch * SEQ * D_MODEL;
  const bf16_t* Kb = Kmat + (long long)batch * SEQ * D_MODEL;

  const int w    = threadIdx.x >> 5;
  const int lane = threadIdx.x & 31;
  const int g    = lane >> 4;
  const int c    = lane & 15;

  // Cache the whole 16 x 512 Q tile as 16 A-fragments (128 VGPRs).
  const bf16_t* aRow = Qb + (long long)(rowInB + c) * D_MODEL;
  Frag16 aF[16];
#pragma unroll
  for (int kf = 0; kf < 16; ++kf) load_afrag(aF[kf], aRow, kf * 32, g);

  for (int t = 0; t < 16; ++t) {
    const int n0 = w * 256 + t * 16;
    const bf16_t* bc = Kb + (long long)(n0 + c) * D_MODEL;
    v8f acc = {};
#pragma unroll
    for (int kf = 0; kf < 16; ++kf) {
      Frag16 bf;
      load_bfrag(bf, bc, kf * 32, g);
      acc = wmma_bf16(aF[kf].v, bf.v, acc);
    }
#pragma unroll
    for (int j = 0; j < 8; ++j) sS[(g * 8 + j) * SEQ + n0 + c] = acc[j];
  }
  __syncthreads();

  // Softmax: 16 threads per row (rows r = tid>>4; sub-groups stay inside a wave).
  const int row = threadIdx.x >> 4;
  const int cc  = threadIdx.x & 15;
  float* srow = sS + row * SEQ;

  float mx = -3.0e38f;
  for (int i = cc; i < SEQ; i += 16) mx = fmaxf(mx, srow[i]);
#pragma unroll
  for (int off = 8; off; off >>= 1) mx = fmaxf(mx, __shfl_xor(mx, off, 16));

  float sum = 0.0f;
  for (int i = cc; i < SEQ; i += 16) {
    float e = __expf(srow[i] - mx);
    srow[i] = e;
    sum += e;
  }
#pragma unroll
  for (int off = 8; off; off >>= 1) sum += __shfl_xor(sum, off, 16);
  const float inv = 1.0f / sum;

  bf16_t* prow = P + (long long)(batch * SEQ + rowInB + row) * SEQ;
  for (int i = cc; i < SEQ; i += 16) prow[i] = __float2bfloat16(srow[i] * inv);
}

// ---------------------------------------------------------------------------
// LayerNorm over last dim (512) -> bf16. One wave per row, 8 rows per block.
// ---------------------------------------------------------------------------
__global__ __launch_bounds__(256) void layernorm_bf16_kernel(
    const float* __restrict__ x, const float* __restrict__ gamma,
    const float* __restrict__ beta, bf16_t* __restrict__ h, int rows) {
  const int w    = threadIdx.x >> 5;
  const int lane = threadIdx.x & 31;
  const int r    = blockIdx.x * 8 + w;
  if (r >= rows) return;
  const float* xr = x + (long long)r * D_MODEL;

  float v[16];
  float s = 0.0f;
#pragma unroll
  for (int i = 0; i < 16; ++i) { v[i] = xr[lane + i * 32]; s += v[i]; }
#pragma unroll
  for (int off = 16; off; off >>= 1) s += __shfl_xor(s, off, 32);
  const float mu = s * (1.0f / (float)D_MODEL);

  float q = 0.0f;
#pragma unroll
  for (int i = 0; i < 16; ++i) { const float d = v[i] - mu; q += d * d; }
#pragma unroll
  for (int off = 16; off; off >>= 1) q += __shfl_xor(q, off, 32);
  const float rs = rsqrtf(q * (1.0f / (float)D_MODEL) + 1e-5f);

  bf16_t* hr = h + (long long)r * D_MODEL;
#pragma unroll
  for (int i = 0; i < 16; ++i) {
    const int idx = lane + i * 32;
    hr[idx] = __float2bfloat16((v[i] - mu) * rs * gamma[idx] + beta[idx]);
  }
}

// fp32 [K,N] -> bf16 transposed [N,K]
__global__ __launch_bounds__(256) void cvt_transpose_kernel(
    const float* __restrict__ in, bf16_t* __restrict__ out, int K, int N) {
  const long long t = (long long)blockIdx.x * 256 + threadIdx.x;
  if (t >= (long long)K * N) return;
  const int k = (int)(t / N);
  const int n = (int)(t % N);
  out[(long long)n * K + k] = __float2bfloat16(in[t]);
}

// ---------------------------------------------------------------------------
extern "C" void kernel_launch(void* const* d_in, const int* in_sizes, int n_in,
                              void* d_out, int out_size, void* d_ws, size_t ws_size,
                              hipStream_t stream) {
  const float* x   = (const float*)d_in[0];
  const float* Wq  = (const float*)d_in[1];
  const float* Wk  = (const float*)d_in[2];
  const float* Wv  = (const float*)d_in[3];
  const float* Wo  = (const float*)d_in[4];
  const float* W1  = (const float*)d_in[5];
  const float* b1  = (const float*)d_in[6];
  const float* W2  = (const float*)d_in[7];
  const float* b2  = (const float*)d_in[8];
  const float* g1  = (const float*)d_in[9];
  const float* be1 = (const float*)d_in[10];
  const float* g2  = (const float*)d_in[11];
  const float* be2 = (const float*)d_in[12];

  // ---- workspace layout (bf16 elements unless noted) -----------------------
  bf16_t* p   = (bf16_t*)d_ws;
  bf16_t* WqT = p; p += (long long)D_MODEL * D_MODEL;      // [512,512]
  bf16_t* WkT = p; p += (long long)D_MODEL * D_MODEL;
  bf16_t* WvT = p; p += (long long)D_MODEL * D_MODEL;
  bf16_t* WoT = p; p += (long long)D_MODEL * D_MODEL;
  bf16_t* W1T = p; p += (long long)D_MODEL * D_FF;         // [2048,512]
  bf16_t* W2T = p; p += (long long)D_FF * D_MODEL;         // [512,2048]
  bf16_t* h   = p; p += (long long)MTOT * D_MODEL;         // LN1 out; reused for LN2
  bf16_t* Qb  = p; p += (long long)MTOT * D_MODEL;         // Q; reused as attn out
  bf16_t* Kb  = p; p += (long long)MTOT * D_MODEL;
  bf16_t* Vt  = p; p += (long long)MTOT * D_MODEL;         // [B][512][2048]
  bf16_t* P   = p; p += (long long)BATCH * SEQ * SEQ;      // probs; reused as GELU out
  float*  x2  = (float*)p;                                 // [16384,512] fp32
  float*  out = (float*)d_out;

  const float qscale = 0.0441941738241592f;  // 1/sqrt(512)

  // 1) weights -> bf16 transposed
  cvt_transpose_kernel<<<(D_MODEL * D_MODEL + 255) / 256, 256, 0, stream>>>(Wq, WqT, D_MODEL, D_MODEL);
  cvt_transpose_kernel<<<(D_MODEL * D_MODEL + 255) / 256, 256, 0, stream>>>(Wk, WkT, D_MODEL, D_MODEL);
  cvt_transpose_kernel<<<(D_MODEL * D_MODEL + 255) / 256, 256, 0, stream>>>(Wv, WvT, D_MODEL, D_MODEL);
  cvt_transpose_kernel<<<(D_MODEL * D_MODEL + 255) / 256, 256, 0, stream>>>(Wo, WoT, D_MODEL, D_MODEL);
  cvt_transpose_kernel<<<(D_MODEL * D_FF + 255) / 256, 256, 0, stream>>>(W1, W1T, D_MODEL, D_FF);
  cvt_transpose_kernel<<<(D_FF * D_MODEL + 255) / 256, 256, 0, stream>>>(W2, W2T, D_FF, D_MODEL);

  // 2) LN1
  layernorm_bf16_kernel<<<MTOT / 8, 256, 0, stream>>>(x, g1, be1, h, MTOT);

  // 3-5) Q (scaled), K, V (transposed store)
  dim3 gQKV(D_MODEL / 256, MTOT / 64, 1);
  gemm_wmma_kernel<false, false, false, 1><<<gQKV, 256, 0, stream>>>(
      h, 0, D_MODEL, WqT, 0, D_MODEL, nullptr, nullptr, Qb, 0, D_MODEL, D_MODEL, qscale);
  gemm_wmma_kernel<false, false, false, 1><<<gQKV, 256, 0, stream>>>(
      h, 0, D_MODEL, WkT, 0, D_MODEL, nullptr, nullptr, Kb, 0, D_MODEL, D_MODEL, 1.0f);
  gemm_wmma_kernel<false, false, false, 2><<<gQKV, 256, 0, stream>>>(
      h, 0, D_MODEL, WvT, 0, D_MODEL, nullptr, nullptr, Vt, 0, D_MODEL, D_MODEL, 1.0f);

  // 6) scores + softmax -> P
  attn_scores_softmax_kernel<<<MTOT / 16, 256, 0, stream>>>(Qb, Kb, P);

  // 7) attn = P @ V  (batched; Vt is already the Bt layout). Write into Qb.
  dim3 gPV(D_MODEL / 256, SEQ / 64, BATCH);
  gemm_wmma_kernel<false, false, false, 1><<<gPV, 256, 0, stream>>>(
      P, (long long)SEQ * SEQ, SEQ, Vt, (long long)D_MODEL * SEQ, SEQ,
      nullptr, nullptr, Qb, (long long)SEQ * D_MODEL, SEQ, D_MODEL, 1.0f);

  // 8) x2 = x + attn @ Wo
  dim3 gO(D_MODEL / 256, MTOT / 64, 1);
  gemm_wmma_kernel<false, true, false, 0><<<gO, 256, 0, stream>>>(
      Qb, 0, D_MODEL, WoT, 0, D_MODEL, nullptr, x, x2, 0, D_MODEL, D_MODEL, 1.0f);

  // LN2 (reuse h)
  layernorm_bf16_kernel<<<MTOT / 8, 256, 0, stream>>>(x2, g2, be2, h, MTOT);

  // 9) g = gelu(h @ W1 + b1)  (reuse P buffer);  out = x2 + g @ W2 + b2
  dim3 gF1(D_FF / 256, MTOT / 64, 1);
  gemm_wmma_kernel<true, false, true, 1><<<gF1, 256, 0, stream>>>(
      h, 0, D_MODEL, W1T, 0, D_MODEL, b1, nullptr, P, 0, D_MODEL, D_FF, 1.0f);
  dim3 gF2(D_MODEL / 256, MTOT / 64, 1);
  gemm_wmma_kernel<true, true, false, 0><<<gF2, 256, 0, stream>>>(
      P, 0, D_FF, W2T, 0, D_FF, b2, x2, out, 0, D_FF, D_MODEL, 1.0f);
}
